// SelfAttention_5660766896667
// MI455X (gfx1250) — matrix operand order repaired and checked by reading, not compiled, not run on hardware
//
#include <hip/hip_runtime.h>

typedef __attribute__((ext_vector_type(16))) _Float16 v16h;
typedef __attribute__((ext_vector_type(8)))  _Float16 v8h;
typedef __attribute__((ext_vector_type(8)))  float    v8f;
typedef __attribute__((ext_vector_type(4)))  float    v4f;

#define LOG2E 1.44269504088896340736f

constexpr int Bn = 2, Sn = 2048, Hn = 16, Dn = 64;
constexpr int QB  = 128;              // queries per workgroup (8 waves x 16)
constexpr int KBL = 32;               // keys per inner block
constexpr int KSTRIDE = 2 * Hn * Dn;  // floats between consecutive s in qk
constexpr int VSTRIDE = Hn * Dn;      // floats between consecutive s in v
constexpr int LP = 72;                // LDS row pitch in halfs (144B: 16B-aligned, bank-friendly)

__device__ __forceinline__ v8f wmma_f16(v16h a, v16h b, v8f c) {
  // D(f32 16x16) = A(f16 16x32) * B(f16 32x16) + C
  return __builtin_amdgcn_wmma_f32_16x16x32_f16(false, a, false, b,
                                                (short)0, c, false, false);
}

__device__ __forceinline__ float bperm(float x, int srcLane) {
  return __builtin_bit_cast(float,
      __builtin_amdgcn_ds_bpermute(srcLane << 2, __builtin_bit_cast(int, x)));
}

#define CAT16(lo, hi) __builtin_shufflevector((lo), (hi), 0, 1, 2, 3, 4, 5, 6, 7, \
                                              8, 9, 10, 11, 12, 13, 14, 15)

__global__ __launch_bounds__(256, 1)
void fa_fwd(const float* __restrict__ qk, const float* __restrict__ vin,
            float* __restrict__ out)
{
  __shared__ __align__(16) _Float16 Ksh[2][KBL][LP];
  __shared__ __align__(16) _Float16 Vsh[2][KBL][LP];

  const int lane = threadIdx.x & 31;
  const int wave = threadIdx.x >> 5;
  const int lq   = lane & 15;      // query column (S^T tiles) / d column (O tiles)
  const int hl   = lane >> 4;      // lane half
  const int h = blockIdx.y;
  const int b = blockIdx.z;
  const int q0  = blockIdx.x * QB + wave * 16;
  const int q_g = q0 + lq;         // this lane's query row (softmax/mask)

  // cooperative staging coordinates: 256 threads cover 32 keys x 64 floats
  const int trow = threadIdx.x >> 3;        // key row within block
  const int tcol = (threadIdx.x & 7) * 8;   // 8-float column group

  const float qscale = 0.125f * LOG2E;      // 1/sqrt(64) * log2(e)
  const float MASKV  = -10000.0f * LOG2E;   // reference's additive mask, exp2 domain

  const float* qbase = qk  + (size_t)(b * Sn * 2 + 0) * (Hn * Dn) + (size_t)h * Dn;
  const float* kbase = qk  + (size_t)(b * Sn * 2 + 1) * (Hn * Dn) + (size_t)h * Dn;
  const float* vbase = vin + (size_t)b * Sn * Hn * Dn + (size_t)h * Dn;

  // ---- Q^T as two WMMA B operands: lane column = query lq, elems = d in table order (0..31 / 32..63)
  v16h qb0, qb1;
  #pragma unroll
  for (int e = 0; e < 16; ++e) {
    const float* qp = qbase + (size_t)(q0 + lq) * KSTRIDE;
    const int kk = (e < 8) ? (hl * 8 + e) : (16 + hl * 8 + e - 8);
    qb0[e] = (_Float16)(qp[kk]      * qscale);
    qb1[e] = (_Float16)(qp[kk + 32] * qscale);
  }

  v8f o0 = {0,0,0,0,0,0,0,0};
  v8f o1 = o0, o2 = o0, o3 = o0;
  float mrow = -3.0e38f;   // running row max (exp2 domain)
  float lrow = 0.0f;       // running row sum

  const int nkb = (blockIdx.x + 1) * (QB / KBL);  // WG-uniform key-block count

  // ---- stage key block 0 into LDS buffer 0 (f32 -> f16 once, shared by 8 waves)
  {
    const float* kp = kbase + (size_t)trow * KSTRIDE + tcol;
    const float* vp = vbase + (size_t)trow * VSTRIDE + tcol;
    v4f a = *(const v4f*)kp, bq = *(const v4f*)(kp + 4);
    v4f c = *(const v4f*)vp, d  = *(const v4f*)(vp + 4);
    v8h kh, vh;
    #pragma unroll
    for (int j = 0; j < 4; ++j) {
      kh[j] = (_Float16)a[j]; kh[4 + j] = (_Float16)bq[j];
      vh[j] = (_Float16)c[j]; vh[4 + j] = (_Float16)d[j];
    }
    *(v8h*)&Ksh[0][trow][tcol] = kh;
    *(v8h*)&Vsh[0][trow][tcol] = vh;
  }
  __syncthreads();

  for (int kb = 0; kb < nkb; ++kb) {
    const int k0  = kb * KBL;
    const int buf = kb & 1;
    const bool have_next = (kb + 1) < nkb;   // WG-uniform

    // ---- software pipeline: issue global loads for block kb+1 now
    v4f nka = {0,0,0,0}, nkc = nka, nva = nka, nvc = nka;
    if (have_next) {
      const float* kp = kbase + (size_t)(k0 + KBL + trow) * KSTRIDE + tcol;
      const float* vp = vbase + (size_t)(k0 + KBL + trow) * VSTRIDE + tcol;
      nka = *(const v4f*)kp; nkc = *(const v4f*)(kp + 4);
      nva = *(const v4f*)vp; nvc = *(const v4f*)(vp + 4);
    }

    // ---- compute (skip blocks entirely above this wave's queries; wave-uniform)
    if (k0 <= q0 + 15) {
      _Float16 (*Ks)[LP] = Ksh[buf];
      _Float16 (*Vs)[LP] = Vsh[buf];

      // S^T = K . Q^T : two 16(key) x 16(query) f32 tiles, accumulate over d
      v8f st0 = {0,0,0,0,0,0,0,0};
      v8f st1 = st0;
      #pragma unroll
      for (int dh = 0; dh < 2; ++dh) {
        const int d0 = dh * 32 + hl * 8;
        // A operand (16 keys x 32 d): lane row = key, elems e<8 -> d0+e, e>=8 -> d0+16+e-8
        v16h ak0 = CAT16(*(const v8h*)&Ks[lq][d0],      *(const v8h*)&Ks[lq][d0 + 16]);
        v16h ak1 = CAT16(*(const v8h*)&Ks[16 + lq][d0], *(const v8h*)&Ks[16 + lq][d0 + 16]);
        const v16h qb = dh ? qb1 : qb0;
        st0 = wmma_f16(ak0, qb, st0);
        st1 = wmma_f16(ak1, qb, st1);
      }

      // causal mask: only the 1-2 diagonal blocks need it (wave-uniform test)
      if (k0 + KBL - 1 > q0) {
        #pragma unroll
        for (int r = 0; r < 8; ++r) {
          const int kg = k0 + hl * 8 + r;
          st0[r] += (kg      > q_g) ? MASKV : 0.0f;
          st1[r] += (kg + 16 > q_g) ? MASKV : 0.0f;
        }
      }

      // online softmax: each lane owns exactly one query
      float smax = -3.0e38f;
      #pragma unroll
      for (int r = 0; r < 8; ++r)
        smax = fmaxf(smax, fmaxf(st0[r], st1[r]));
      smax = fmaxf(smax, bperm(smax, lane ^ 16));
      const float m_new = fmaxf(mrow, smax);
      const float alpha = __builtin_amdgcn_exp2f(mrow - m_new);
      float lsum = 0.0f;
      v16h pa;   // P directly in A-operand layout: elems 0..7 <- st0, 8..15 <- st1
      #pragma unroll
      for (int r = 0; r < 8; ++r) {
        const float e0 = __builtin_amdgcn_exp2f(st0[r] - m_new);
        const float e1 = __builtin_amdgcn_exp2f(st1[r] - m_new);
        lsum += e0 + e1;
        pa[r]     = (_Float16)e0;
        pa[8 + r] = (_Float16)e1;
      }
      lsum += bperm(lsum, lane ^ 16);
      lrow = lrow * alpha + lsum;
      mrow = m_new;

      // V as four B operands (32 keys x 16 d): lane column = d lq, elems = keys in table order
      v16h vb[4];
      #pragma unroll
      for (int j = 0; j < 4; ++j)
        #pragma unroll
        for (int e = 0; e < 16; ++e)
          vb[j][e] = Vs[(e < 8) ? (hl * 8 + e) : (16 + hl * 8 + e - 8)][16 * j + lq];

      // rescale O by alpha(query-row) broadcast from the softmax lanes
      #pragma unroll
      for (int r = 0; r < 8; ++r) {
        const float ar = bperm(alpha, hl * 8 + r);
        o0[r] *= ar; o1[r] *= ar; o2[r] *= ar; o3[r] *= ar;
      }
      o0 = wmma_f16(pa, vb[0], o0);
      o1 = wmma_f16(pa, vb[1], o1);
      o2 = wmma_f16(pa, vb[2], o2);
      o3 = wmma_f16(pa, vb[3], o3);
    }

    // ---- convert & store next block into the other LDS buffer
    if (have_next) {
      v8h kh, vh;
      #pragma unroll
      for (int j = 0; j < 4; ++j) {
        kh[j] = (_Float16)nka[j]; kh[4 + j] = (_Float16)nkc[j];
        vh[j] = (_Float16)nva[j]; vh[4 + j] = (_Float16)nvc[j];
      }
      *(v8h*)&Ksh[buf ^ 1][trow][tcol] = kh;
      *(v8h*)&Vsh[buf ^ 1][trow][tcol] = vh;
    }
    __syncthreads();   // orders: stores(buf^1) before next reads; reads(buf) before overwrite
  }

  // ---- normalize by row sum and store: out[b, q, h, d]
  #pragma unroll
  for (int r = 0; r < 8; ++r) {
    const float lr  = bperm(lrow, hl * 8 + r);
    const float inv = __builtin_amdgcn_rcpf(lr);
    const int s = q0 + hl * 8 + r;
    float* op = out + (((size_t)b * Sn + s) * Hn + h) * Dn + lq;
    op[0]  = o0[r] * inv;
    op[16] = o1[r] * inv;
    op[32] = o2[r] * inv;
    op[48] = o3[r] * inv;
  }
}

extern "C" void kernel_launch(void* const* d_in, const int* in_sizes, int n_in,
                              void* d_out, int out_size, void* d_ws, size_t ws_size,
                              hipStream_t stream) {
  (void)in_sizes; (void)n_in; (void)out_size; (void)d_ws; (void)ws_size;
  const float* qk = (const float*)d_in[0];
  const float* v  = (const float*)d_in[1];
  float* out      = (float*)d_out;
  dim3 grid(Sn / QB, Hn, Bn);   // (16, 16, 2)
  dim3 block(256, 1, 1);        // 8 wave32 per workgroup
  hipLaunchKernelGGL(fa_fwd, grid, block, 0, stream, qk, v, out);
}
